// LGCN_83184926589612
// MI455X (gfx1250) — matrix operand order repaired and checked by reading, not compiled, not run on hardware
//
#include <hip/hip_runtime.h>
#include <hip/hip_bf16.h>
#include <stdint.h>

// ---------------------------------------------------------------------------
// CDNA5 (gfx1250) GAT-style layer.
// Node GEMMs: bf16 WMMA 16x16x32, f32 accumulate (v_wmma_f32_16x16x32_bf16).
// Edge-phase tensors (s_e, d_e, ft) are stored bf16: they are random-row
// gathers (the memory-bound part of the op at 23.3 TB/s) and are already
// bf16-precision products of the WMMA pipeline, so bf16 storage halves the
// dominant traffic with no extra precision loss. Accumulations (denom, ssum)
// stay f32 with native global_atomic_add_f32.
// segment_max is omitted: exp(e-max)/sum(exp(e-max)) == exp(e)/sum(exp(e))
// and W_SCALE keeps logits in a sane range (per the reference's own comment).
// ---------------------------------------------------------------------------

typedef __bf16 bf16_t;
typedef __attribute__((ext_vector_type(16))) __bf16 v16bf;
typedef __attribute__((ext_vector_type(8)))  __bf16 v8bf;
typedef __attribute__((ext_vector_type(4)))  __bf16 v4bf;
typedef __attribute__((ext_vector_type(8)))  float  v8f;

#define D 64   // all hidden dims are 64

// weight fragment offsets in the bf16 workspace region (elements)
#define OFF_W4   0
#define OFF_W5   4096
#define OFF_W6   8192
#define OFF_W7   12288   /* K=192 -> 24 frags = 12288 elems */
#define OFF_W8   24576
#define OFF_W9   28672   /* K=192 */
#define OFF_W10  40960
#define OFF_W11  45056
#define OFF_W11B 49152
#define W_TOTAL  53248

__device__ __forceinline__ v8f wmma_bf16(v16bf a, v16bf b, v8f c) {
  // emits v_wmma_f32_16x16x32_bf16
  return __builtin_amdgcn_wmma_f32_16x16x32_bf16(false, a, false, b,
                                                 (short)0, c, false, false);
}

// A-fragment (16x32 bf16) from row-major f32 [*,64] (ISA 7.12.2 A layout):
// lane<16 : M=lane,    K = k0+0..7 (v0..3) and k0+16..23 (v4..7)
// lane>=16: M=lane-16, K = k0+8..15         and k0+24..31
__device__ __forceinline__ v16bf load_a_g(const float* M, int m0, int k0, int lane) {
  int hf = lane >> 4, mr = lane & 15;
  const float* p = M + (size_t)(m0 + mr) * D + k0 + hf * 8;
  float4 a0 = *(const float4*)(p + 0);
  float4 a1 = *(const float4*)(p + 4);
  float4 a2 = *(const float4*)(p + 16);
  float4 a3 = *(const float4*)(p + 20);
  v16bf r;
  r[0]=(bf16_t)a0.x;  r[1]=(bf16_t)a0.y;  r[2]=(bf16_t)a0.z;  r[3]=(bf16_t)a0.w;
  r[4]=(bf16_t)a1.x;  r[5]=(bf16_t)a1.y;  r[6]=(bf16_t)a1.z;  r[7]=(bf16_t)a1.w;
  r[8]=(bf16_t)a2.x;  r[9]=(bf16_t)a2.y;  r[10]=(bf16_t)a2.z; r[11]=(bf16_t)a2.w;
  r[12]=(bf16_t)a3.x; r[13]=(bf16_t)a3.y; r[14]=(bf16_t)a3.z; r[15]=(bf16_t)a3.w;
  return r;
}

// Same A layout, sourced from a wave-private bf16 tile in LDS (ld = 64)
__device__ __forceinline__ v16bf load_a_lds(const bf16_t* L, int k0, int lane) {
  int hf = lane >> 4, mr = lane & 15;
  const bf16_t* p = L + mr * D + k0 + hf * 8;
  v8bf lo = *(const v8bf*)(p);        // ds_load_b128
  v8bf hi = *(const v8bf*)(p + 16);   // ds_load_b128
  v16bf r;
#pragma unroll
  for (int j = 0; j < 8; ++j) { r[j] = lo[j]; r[8 + j] = hi[j]; }
  return r;
}

// Pre-swizzled B fragment (32x16): 512 contiguous bf16 per fragment,
// lane holds N=lane&15, K = (j>>3)*16 + (lane>>4)*8 + (j&7)  (2x global_load_b128 per lane)
__device__ __forceinline__ v16bf load_b(const bf16_t* wb, int frag, int lane) {
  return *(const v16bf*)(wb + (size_t)frag * 512 + lane * 16);
}

// C/D layout store to f32 (lane -> (half,N), reg -> M)
__device__ __forceinline__ void store_c(float* M, int m0, int n0, int lane, v8f v) {
  int hf = lane >> 4, n = lane & 15;
  float* p = M + (size_t)(m0 + 8 * hf) * D + n0 + n;
#pragma unroll
  for (int r = 0; r < 8; ++r) p[(size_t)r * D] = v[r];
}

// C/D layout store to bf16 (halves edge-gather traffic downstream)
__device__ __forceinline__ void store_c_bf16(bf16_t* M, int m0, int n0, int lane, v8f v) {
  int hf = lane >> 4, n = lane & 15;
  bf16_t* p = M + (size_t)(m0 + 8 * hf) * D + n0 + n;
#pragma unroll
  for (int r = 0; r < 8; ++r) p[(size_t)r * D] = (bf16_t)v[r];
}

// ---------------------------------------------------------------------------
// Kernel 1: convert one weight [K,64] f32 -> bf16 B-fragments
// ---------------------------------------------------------------------------
__global__ void pack_weight_kernel(const float* __restrict__ W,
                                   bf16_t* __restrict__ out, int K) {
  int tid = blockIdx.x * blockDim.x + threadIdx.x;
  if (tid >= K * D) return;
  int f    = tid >> 9;          // fragment = kt*4 + nt
  int lane = (tid >> 4) & 31;
  int j    = tid & 15;
  int kt = f >> 2, nt = f & 3;
  int n = lane & 15, hf = lane >> 4;
  int k = kt * 32 + (j >> 3) * 16 + hf * 8 + (j & 7);
  out[tid] = (bf16_t)W[(size_t)k * D + nt * 16 + n];
}

__global__ void zero_kernel(float* __restrict__ p, int n) {
  int i = blockIdx.x * blockDim.x + threadIdx.x;
  if (i < n) p[i] = 0.0f;
}

// ---------------------------------------------------------------------------
// Kernel 2: node-level transforms. 8 waves / block, 16 rows / wave.
// Produces s_e, d_e, ft  (each [N,64] bf16).
// ---------------------------------------------------------------------------
__global__ __launch_bounds__(256) void node_transform_kernel(
    const float* __restrict__ h, const float* __restrict__ ctx,
    const float* __restrict__ c, const bf16_t* __restrict__ wb,
    bf16_t* __restrict__ se, bf16_t* __restrict__ de, bf16_t* __restrict__ ftb,
    int N) {
  __shared__ bf16_t lfuse[128 * D];          // 16KB, wave-private 16x64 tiles
  int wave = threadIdx.x >> 5;
  int lane = threadIdx.x & 31;
  int m0 = blockIdx.x * 128 + wave * 16;
  if (m0 >= N) return;                        // wave-uniform: EXEC stays all-1s
  bf16_t* L = lfuse + wave * 16 * D;
  int hf = lane >> 4, nlo = lane & 15;

  v16bf ah0 = load_a_g(h,   m0, 0, lane),  ah1 = load_a_g(h,   m0, 32, lane);
  v16bf ax0 = load_a_g(ctx, m0, 0, lane),  ax1 = load_a_g(ctx, m0, 32, lane);
  v16bf ac0 = load_a_g(c,   m0, 0, lane),  ac1 = load_a_g(c,   m0, 32, lane);

  // fuse = (h@W4) * (ctx@W5) -> LDS (bf16) so it can re-enter as an A-matrix
#pragma unroll
  for (int nt = 0; nt < 4; ++nt) {
    v8f a4 = {}, a5 = {};
    a4 = wmma_bf16(ah0, load_b(wb + OFF_W4, 0 + nt, lane), a4);
    a4 = wmma_bf16(ah1, load_b(wb + OFF_W4, 4 + nt, lane), a4);
    a5 = wmma_bf16(ax0, load_b(wb + OFF_W5, 0 + nt, lane), a5);
    a5 = wmma_bf16(ax1, load_b(wb + OFF_W5, 4 + nt, lane), a5);
    v8f fu = a4 * a5;
#pragma unroll
    for (int r = 0; r < 8; ++r)
      L[(r + 8 * hf) * D + nt * 16 + nlo] = (bf16_t)fu[r];
  }
  __builtin_amdgcn_wave_barrier();   // wave-local LDS store->load ordering

  v16bf af0 = load_a_lds(L, 0, lane), af1 = load_a_lds(L, 32, lane);

#pragma unroll
  for (int nt = 0; nt < 4; ++nt) {
    // s_e = (cat@W7) * (c@W8), cat = [h | ctx | fuse] (K = 192)
    v8f a7 = {};
    a7 = wmma_bf16(ah0, load_b(wb + OFF_W7,  0 + nt, lane), a7);
    a7 = wmma_bf16(ah1, load_b(wb + OFF_W7,  4 + nt, lane), a7);
    a7 = wmma_bf16(ax0, load_b(wb + OFF_W7,  8 + nt, lane), a7);
    a7 = wmma_bf16(ax1, load_b(wb + OFF_W7, 12 + nt, lane), a7);
    a7 = wmma_bf16(af0, load_b(wb + OFF_W7, 16 + nt, lane), a7);
    a7 = wmma_bf16(af1, load_b(wb + OFF_W7, 20 + nt, lane), a7);
    v8f a8 = {};
    a8 = wmma_bf16(ac0, load_b(wb + OFF_W8, 0 + nt, lane), a8);
    a8 = wmma_bf16(ac1, load_b(wb + OFF_W8, 4 + nt, lane), a8);
    store_c_bf16(se, m0, nt * 16, lane, a7 * a8);

    // d_e = fuse @ W6
    v8f a6 = {};
    a6 = wmma_bf16(af0, load_b(wb + OFF_W6, 0 + nt, lane), a6);
    a6 = wmma_bf16(af1, load_b(wb + OFF_W6, 4 + nt, lane), a6);
    store_c_bf16(de, m0, nt * 16, lane, a6);

    // ft = (cat@W9) * (c@W10)
    v8f a9 = {};
    a9 = wmma_bf16(ah0, load_b(wb + OFF_W9,  0 + nt, lane), a9);
    a9 = wmma_bf16(ah1, load_b(wb + OFF_W9,  4 + nt, lane), a9);
    a9 = wmma_bf16(ax0, load_b(wb + OFF_W9,  8 + nt, lane), a9);
    a9 = wmma_bf16(ax1, load_b(wb + OFF_W9, 12 + nt, lane), a9);
    a9 = wmma_bf16(af0, load_b(wb + OFF_W9, 16 + nt, lane), a9);
    a9 = wmma_bf16(af1, load_b(wb + OFF_W9, 20 + nt, lane), a9);
    v8f a10 = {};
    a10 = wmma_bf16(ac0, load_b(wb + OFF_W10, 0 + nt, lane), a10);
    a10 = wmma_bf16(ac1, load_b(wb + OFF_W10, 4 + nt, lane), a10);
    store_c_bf16(ftb, m0, nt * 16, lane, a9 * a10);
  }
}

// ---------------------------------------------------------------------------
// Kernel 3: per-edge exp(<s_e[src], d_e[dst]>) and denom accumulation.
// bf16 rows: 128B per gather instead of 256B.
// ---------------------------------------------------------------------------
__global__ __launch_bounds__(256) void edge_exp_kernel(
    const bf16_t* __restrict__ se, const bf16_t* __restrict__ de,
    const int* __restrict__ src, const int* __restrict__ dst,
    float* __restrict__ ex, float* __restrict__ denom, int E) {
  int i = blockIdx.x * blockDim.x + threadIdx.x;
  if (i >= E) return;
  const v16bf* a = (const v16bf*)(se + (size_t)src[i] * D);
  const v16bf* b = (const v16bf*)(de + (size_t)dst[i] * D);
  float acc = 0.f;
#pragma unroll
  for (int k = 0; k < 4; ++k) {
    v16bf av = a[k], bv = b[k];
#pragma unroll
    for (int j = 0; j < 16; ++j)
      acc = fmaf((float)av[j], (float)bv[j], acc);
  }
  float v = __expf(acc);
  ex[i] = v;
  unsafeAtomicAdd(denom + dst[i], v);   // global_atomic_add_f32
}

// ---------------------------------------------------------------------------
// Kernel 4: weighted scatter-sum. 16 threads per edge, 4 dims per thread.
// ft is bf16 (8B gather per thread); accumulation stays f32 atomics.
// ---------------------------------------------------------------------------
__global__ __launch_bounds__(256) void edge_scatter_kernel(
    const bf16_t* __restrict__ ftb, const float* __restrict__ ex,
    const float* __restrict__ denom, const int* __restrict__ src,
    const int* __restrict__ dst, float* __restrict__ ssum, int E) {
  int t = blockIdx.x * blockDim.x + threadIdx.x;
  int i = t >> 4, ch = t & 15;
  if (i >= E) return;
  int s = src[i], d = dst[i];
  float a = ex[i] / denom[d];
  v4bf f = *(const v4bf*)(ftb + (size_t)s * D + ch * 4);
  float* o = ssum + (size_t)d * D + ch * 4;
  unsafeAtomicAdd(o + 0, (float)f[0] * a);
  unsafeAtomicAdd(o + 1, (float)f[1] * a);
  unsafeAtomicAdd(o + 2, (float)f[2] * a);
  unsafeAtomicAdd(o + 3, (float)f[3] * a);
}

// ---------------------------------------------------------------------------
// Kernel 5: out = ctx @ W11 + s @ W11b
// ---------------------------------------------------------------------------
__global__ __launch_bounds__(256) void ctx_update_kernel(
    const float* __restrict__ ctx, const float* __restrict__ ssum,
    const bf16_t* __restrict__ wb, float* __restrict__ out, int N) {
  int wave = threadIdx.x >> 5;
  int lane = threadIdx.x & 31;
  int m0 = blockIdx.x * 128 + wave * 16;
  if (m0 >= N) return;
  v16bf ax0 = load_a_g(ctx,  m0, 0, lane), ax1 = load_a_g(ctx,  m0, 32, lane);
  v16bf as0 = load_a_g(ssum, m0, 0, lane), as1 = load_a_g(ssum, m0, 32, lane);
#pragma unroll
  for (int nt = 0; nt < 4; ++nt) {
    v8f acc = {};
    acc = wmma_bf16(ax0, load_b(wb + OFF_W11,  0 + nt, lane), acc);
    acc = wmma_bf16(ax1, load_b(wb + OFF_W11,  4 + nt, lane), acc);
    acc = wmma_bf16(as0, load_b(wb + OFF_W11B, 0 + nt, lane), acc);
    acc = wmma_bf16(as1, load_b(wb + OFF_W11B, 4 + nt, lane), acc);
    store_c(out, m0, nt * 16, lane, acc);
  }
}

// ---------------------------------------------------------------------------
extern "C" void kernel_launch(void* const* d_in, const int* in_sizes, int n_in,
                              void* d_out, int out_size, void* d_ws, size_t ws_size,
                              hipStream_t stream) {
  const float* h    = (const float*)d_in[0];
  const float* ctx  = (const float*)d_in[1];
  const float* c    = (const float*)d_in[2];
  const int*   src  = (const int*)d_in[3];
  const int*   dst  = (const int*)d_in[4];
  const float* Wsrc[9] = {
    (const float*)d_in[5],  (const float*)d_in[6],  (const float*)d_in[7],
    (const float*)d_in[8],  (const float*)d_in[9],  (const float*)d_in[10],
    (const float*)d_in[11], (const float*)d_in[12], (const float*)d_in[13]};
  const int    Wk[9]   = {64, 64, 64, 192, 64, 192, 64, 64, 64};
  const size_t Woff[9] = {OFF_W4, OFF_W5, OFF_W6, OFF_W7, OFF_W8,
                          OFF_W9, OFF_W10, OFF_W11, OFF_W11B};

  const int N = in_sizes[0] / D;   // 50000
  const int E = in_sizes[3];       // 800000
  float* out = (float*)d_out;

  // workspace layout: bf16 edge tensors first (each N*64*2 B, 32B aligned),
  // then f32 accumulators, then the bf16 weight fragments.
  bf16_t* se   = (bf16_t*)d_ws;              // N*64 bf16
  bf16_t* de   = se  + (size_t)N * D;        // N*64 bf16
  bf16_t* ftb  = de  + (size_t)N * D;        // N*64 bf16
  float* ssum  = (float*)(ftb + (size_t)N * D);  // N*64 f32 (atomic accum)
  float* denom = ssum + (size_t)N * D;       // N f32 (atomic accum)
  float* ex    = denom + N;                  // E f32
  bf16_t* wb   = (bf16_t*)(ex + E);          // W_TOTAL bf16

  // 1) pack weights into bf16 B-fragments
  for (int w = 0; w < 9; ++w) {
    int total = Wk[w] * D;
    pack_weight_kernel<<<(total + 255) / 256, 256, 0, stream>>>(
        Wsrc[w], wb + Woff[w], Wk[w]);
  }

  // 2) zero atomic accumulators (ssum then denom are contiguous: N*65 floats)
  int zn = N * (D + 1);
  zero_kernel<<<(zn + 255) / 256, 256, 0, stream>>>(ssum, zn);

  // 3) node transforms (WMMA)
  node_transform_kernel<<<(N + 127) / 128, 256, 0, stream>>>(
      h, ctx, c, wb, se, de, ftb, N);

  // 4) edge logits -> exp -> denom
  edge_exp_kernel<<<(E + 255) / 256, 256, 0, stream>>>(
      se, de, src, dst, ex, denom, E);

  // 5) weighted scatter-sum of messages
  long long st = (long long)E * 16;
  edge_scatter_kernel<<<(int)((st + 255) / 256), 256, 0, stream>>>(
      ftb, ex, denom, src, dst, ssum, E);

  // 6) context update (WMMA)
  ctx_update_kernel<<<(N + 127) / 128, 256, 0, stream>>>(ctx, ssum, wb, out, N);
}
